// StackRNN_35828617183557
// MI455X (gfx1250) — matrix operand order repaired and checked by reading, not compiled, not run on hardware
//
#include <hip/hip_runtime.h>
#include <cstdint>
#include <cstddef>

// Problem constants (from reference)
#define SEQ   200
#define BATCH 256
#define INDIM 256
#define HDIM  512
#define SWDIM 64
#define DEPTH 100
#define KTOT  832   // INDIM + SWDIM + HDIM (concatenated K for fused GEMM)
#define G4H   2048  // 4*HDIM
#define NKK   (KTOT/32)   // 26 K-steps of 32

typedef __attribute__((ext_vector_type(16))) __bf16 v16bf;
typedef __attribute__((ext_vector_type(8)))  float  v8f;

union B16x16 { uint32_t u[8]; v16bf v; };

__device__ __forceinline__ float sigm(float x){ return 1.0f/(1.0f+__expf(-x)); }

// ---------------- one-time init kernels ----------------

// Pack [W_ih | W_hh] -> bf16 row-major [2048 x 832]; biasSum = b_ih + b_hh.
__global__ void pack_weights(const float* __restrict__ Wih, const float* __restrict__ Whh,
                             const float* __restrict__ bih, const float* __restrict__ bhh,
                             __bf16* __restrict__ Wcat, float* __restrict__ biasSum){
  int n = blockIdx.x;                       // 0..2047 (gate-unit row)
  const float* wi = Wih + (size_t)n*(INDIM+SWDIM);
  const float* wh = Whh + (size_t)n*HDIM;
  __bf16* dst = Wcat + (size_t)n*KTOT;
  for (int k = threadIdx.x; k < KTOT; k += blockDim.x){
    float v = (k < INDIM+SWDIM) ? wi[k] : wh[k-(INDIM+SWDIM)];
    dst[k] = (__bf16)v;
  }
  if (threadIdx.x == 0) biasSum[n] = bih[n] + bhh[n];
}

// inp[b][0:256] = bf16(x[0][b]); inp[b][256:832] = 0 (stack top & h start at zero).
// Also zero c and stack ping buffer.
__global__ void init_state(const float* __restrict__ x, __bf16* __restrict__ inp,
                           float* __restrict__ cbuf, float* __restrict__ stack0){
  int i = blockIdx.x*blockDim.x + threadIdx.x;
  int stride = gridDim.x*blockDim.x;
  for (int k = i; k < BATCH*KTOT; k += stride){
    int b = k / KTOT, c = k - b*KTOT;
    inp[k] = (c < INDIM) ? (__bf16)x[(size_t)b*INDIM + c] : (__bf16)0.f;
  }
  for (int k = i; k < BATCH*HDIM;        k += stride) cbuf[k]   = 0.f;
  for (int k = i; k < BATCH*DEPTH*SWDIM; k += stride) stack0[k] = 0.f;
}

// ---------------- per-step kernel R: fused gates GEMM + LSTM pointwise ----------------
// grid = 64 blocks x 256 threads (8 waves). blockIdx low 4 bits -> M tile (16 rows of B=256),
// high 2 bits -> 128-wide N group; wave -> 16-wide N tile within [0,512).
// Each wave computes the 4 gate tiles (N, N+512, N+1024, N+1536) sharing one A operand.
// A operand (16x32 bf16 tile of inp[B x 832]) is double-buffered in LDS and fetched with
// GLOBAL_LOAD_ASYNC_TO_LDS_B64 so tile kk+1 streams in while tile kk feeds the WMMAs.
__global__ void __launch_bounds__(256, 2)
lstm_step_gemm(const __bf16* __restrict__ inp, const __bf16* __restrict__ Wcat,
               const float* __restrict__ biasSum, float* __restrict__ cbuf,
               float* __restrict__ out, int t)
{
  __shared__ alignas(16) __bf16 ldsA[2][16*32];   // double-buffered 16(M) x 32(K) bf16 tile

  const int tid  = threadIdx.x;
  const int lane = tid & 31;
  const int wave = tid >> 5;
  const int hi   = lane >> 4;     // lane group (0/1)
  const int lm   = lane & 15;
  const int m0   = (blockIdx.x & 15) << 4;                 // batch-row tile base
  const int n0   = ((blockIdx.x >> 4) << 7) + (wave << 4); // h-column tile base, in [0,512)

  v8f acc[4] = {};   // i, f, g, o accumulators (fp32)

  // B-operand row pointers (dword view) for the 4 gates: row = g*512 + n0 + lm of Wcat[2048][832]
  const uint32_t* brow[4];
#pragma unroll
  for (int g = 0; g < 4; ++g)
    brow[g] = (const uint32_t*)(Wcat + (size_t)(g*HDIM + n0 + lm)*KTOT);

  // Async A-tile copy: waves 0..3, each lane moves one 8-byte chunk.
  // lane -> row 4*wave + (lane>>3) of the tile, halfs [4*(lane&7) .. +4)
  unsigned long long gsrc = 0;
  uint32_t ldst0 = 0, ldst1 = 0;
  if (wave < 4){
    int rowL  = (wave << 2) + (lane >> 3);     // 0..15
    int chunk = lane & 7;                      // 8 chunks of 4 halfs
    gsrc  = (unsigned long long)(uintptr_t)(inp + (size_t)(m0 + rowL)*KTOT + (chunk << 2));
    // flat LDS address low 32 bits == LDS byte offset (flat LDS aperture)
    ldst0 = (uint32_t)(uintptr_t)&ldsA[0][rowL*32 + (chunk << 2)];
    ldst1 = (uint32_t)(uintptr_t)&ldsA[1][rowL*32 + (chunk << 2)];
    // prime tile 0
    asm volatile("global_load_async_to_lds_b64 %0, %1, off"
                 :: "v"(ldst0), "v"(gsrc) : "memory");
    gsrc += 64;   // 32 halfs per K-step
  }

  for (int kk = 0; kk < NKK; ++kk){
    if (wave < 4)
      asm volatile("s_wait_asynccnt 0x0" ::: "memory");   // tile kk landed (per-wave counter)
    __syncthreads();                                      // visible to all waves; prior reads done
    if (wave < 4 && kk + 1 < NKK){
      uint32_t dst = ((kk + 1) & 1) ? ldst1 : ldst0;
      asm volatile("global_load_async_to_lds_b64 %0, %1, off"
                   :: "v"(dst), "v"(gsrc) : "memory");    // prefetch tile kk+1
      gsrc += 64;
    }

    const uint32_t* a32 = (const uint32_t*)&ldsA[kk & 1][0];

    // A operand per ISA layout: lane m=lm, halves 2v,2v+1 -> K = (v<4?0:16)+hi*8+(v&3)*2 (+1)
    B16x16 a;
#pragma unroll
    for (int v = 0; v < 8; ++v){
      int dk = ((v < 4) ? 0 : 8) + hi*4 + (v & 3);   // dword index within the 16-dword LDS row
      a.u[v] = a32[lm*16 + dk];
    }

    // B operand: lane n=lm, halves 2v,2v+1 -> K = (v<4?0:16)+hi*8+(v&3)*2 (+1), same as A
    const int kd = kk << 4;   // (kk*32)/2
#pragma unroll
    for (int g = 0; g < 4; ++g){
      B16x16 b;
#pragma unroll
      for (int v = 0; v < 8; ++v) b.u[v] = brow[g][kd + ((v < 4) ? 0 : 8) + hi*4 + (v & 3)];
      acc[g] = __builtin_amdgcn_wmma_f32_16x16x32_bf16(
                 false, a.v, false, b.v, (short)0, acc[g], false, false);
    }
  }

  // Fused LSTM pointwise on the C/D layout: lane col N=lm, VGPR r -> row M = r + 8*hi.
  const int h  = n0 + lm;
  const float bi = biasSum[h];
  const float bf = biasSum[h + HDIM];
  const float bg = biasSum[h + 2*HDIM];
  const float bo = biasSum[h + 3*HDIM];
  const size_t obase = (size_t)t*BATCH*HDIM;
#pragma unroll
  for (int r = 0; r < 8; ++r){
    int bidx = m0 + r + hi*8;
    float iv = acc[0][r] + bi;
    float fv = acc[1][r] + bf;
    float gv = acc[2][r] + bg;
    float ov = acc[3][r] + bo;
    size_t ci = (size_t)bidx*HDIM + h;
    float cp = cbuf[ci];
    float cn = sigm(fv)*cp + sigm(iv)*tanhf(gv);
    float hn = sigm(ov)*tanhf(cn);
    cbuf[ci]        = cn;
    out[obase + ci] = hn;
  }
}

// ---------------- per-step kernel S: controls + stack_input + stack shift ----------------
// One block per batch row; h row and the 100x64 stack staged in LDS.
// Also assembles next step's A-operand row: inp[b] = [x_{t+1}[b] | new_top[b] | h_t[b]] (bf16).
__global__ void __launch_bounds__(256, 2)
stack_step(const float* __restrict__ out, const float* __restrict__ x,
           const float* __restrict__ Aw, const float* __restrict__ Ab,
           const float* __restrict__ Dw, const float* __restrict__ Db,
           const float* __restrict__ stack_in, float* __restrict__ stack_out,
           __bf16* __restrict__ inp, int t)
{
  __shared__ float lh[HDIM];
  __shared__ float lsi[SWDIM];
  __shared__ float llog[3];
  __shared__ float lstk[DEPTH*SWDIM];

  const int b   = blockIdx.x;
  const int tid = threadIdx.x;
  const float* hrow  = out + (size_t)t*BATCH*HDIM + (size_t)b*HDIM;
  const float* sbase = stack_in + (size_t)b*DEPTH*SWDIM;
  __bf16* irow = inp + (size_t)b*KTOT;

  for (int k = tid; k < HDIM;        k += 256) lh[k]   = hrow[k];
  for (int k = tid; k < DEPTH*SWDIM; k += 256) lstk[k] = sbase[k];
  __syncthreads();

  // Assemble inp for step t+1: h part and x part (stack-top part written below).
  for (int k = tid; k < HDIM; k += 256) irow[INDIM + SWDIM + k] = (__bf16)lh[k];
  if (t + 1 < SEQ){
    const float* xrow = x + (size_t)(t + 1)*BATCH*INDIM + (size_t)b*INDIM;
    for (int k = tid; k < INDIM; k += 256) irow[k] = (__bf16)xrow[k];
  }

  if (tid < SWDIM){                       // stack_input[j] = tanh(D_w[j,:] . h + D_b[j])
    const float* w = Dw + (size_t)tid*HDIM;
    float s = Db[tid];
    for (int k = 0; k < HDIM; ++k) s = fmaf(w[k], lh[k], s);
    lsi[tid] = tanhf(s);
  } else if (tid < SWDIM + 3){            // control logits (push, pop, noop)
    int c = tid - SWDIM;
    const float* w = Aw + (size_t)c*HDIM;
    float s = Ab[c];
    for (int k = 0; k < HDIM; ++k) s = fmaf(w[k], lh[k], s);
    llog[c] = s;
  }
  __syncthreads();

  float l0 = llog[0], l1 = llog[1], l2 = llog[2];
  float mx = fmaxf(l0, fmaxf(l1, l2));
  float e0 = __expf(l0-mx), e1 = __expf(l1-mx), e2 = __expf(l2-mx);
  float inv = 1.0f/(e0+e1+e2);
  float a_push = e0*inv, a_pop = e1*inv, a_noop = e2*inv;

  float* obase = stack_out + (size_t)b*DEPTH*SWDIM;
  for (int e = tid; e < DEPTH*SWDIM; e += 256){
    int d = e >> 6, j = e & 63;
    float up   = d ? lstk[e - SWDIM] : lsi[j];
    float down = (d == DEPTH-1) ? 0.f : lstk[e + SWDIM];
    float v = a_noop*lstk[e] + a_push*up + a_pop*down;
    obase[e] = v;
    if (d == 0) irow[INDIM + j] = (__bf16)v;   // bf16 stack-top for next step's A operand
  }
}

// ---------------- host entry ----------------
extern "C" void kernel_launch(void* const* d_in, const int* in_sizes, int n_in,
                              void* d_out, int out_size, void* d_ws, size_t ws_size,
                              hipStream_t stream){
  (void)in_sizes; (void)n_in; (void)out_size; (void)ws_size;
  const float* x   = (const float*)d_in[0];
  const float* Wih = (const float*)d_in[1];
  const float* Whh = (const float*)d_in[2];
  const float* bih = (const float*)d_in[3];
  const float* bhh = (const float*)d_in[4];
  const float* Aw  = (const float*)d_in[5];
  const float* Ab  = (const float*)d_in[6];
  const float* Dw  = (const float*)d_in[7];
  const float* Db  = (const float*)d_in[8];
  float* out = (float*)d_out;

  // Workspace carve-out (~18 MB; everything stays L2-resident on MI455X's 192 MB L2)
  char* ws = (char*)d_ws;
  size_t o = 0;
  auto alloc = [&](size_t bytes){ void* p = ws + o; o += (bytes + 255) & ~(size_t)255; return p; };
  __bf16* Wcat    = (__bf16*)alloc((size_t)G4H*KTOT*2);
  float*  biasSum = (float*) alloc((size_t)G4H*4);
  __bf16* inp     = (__bf16*)alloc((size_t)BATCH*KTOT*2);
  float*  cbuf    = (float*) alloc((size_t)BATCH*HDIM*4);
  float*  stk0    = (float*) alloc((size_t)BATCH*DEPTH*SWDIM*4);
  float*  stk1    = (float*) alloc((size_t)BATCH*DEPTH*SWDIM*4);
  float*  stk[2]  = { stk0, stk1 };

  pack_weights<<<G4H, 256, 0, stream>>>(Wih, Whh, bih, bhh, Wcat, biasSum);
  init_state<<<512, 256, 0, stream>>>(x, inp, cbuf, stk[0]);

  for (int t = 0; t < SEQ; ++t){
    lstm_step_gemm<<<64, 256, 0, stream>>>(inp, Wcat, biasSum, cbuf, out, t);
    stack_step<<<BATCH, 256, 0, stream>>>(out, x, Aw, Ab, Dw, Db,
                                          stk[t & 1], stk[(t + 1) & 1], inp, t);
  }

  // h_n = outputs[SEQ-1]; c_n = final cell state.
  float* hn = out + (size_t)SEQ*BATCH*HDIM;
  float* cn = hn + (size_t)BATCH*HDIM;
  hipMemcpyAsync(hn, out + (size_t)(SEQ-1)*BATCH*HDIM, (size_t)BATCH*HDIM*sizeof(float),
                 hipMemcpyDeviceToDevice, stream);
  hipMemcpyAsync(cn, cbuf, (size_t)BATCH*HDIM*sizeof(float),
                 hipMemcpyDeviceToDevice, stream);
}
